// UnifiedInteractionLayer_57638461112854
// MI455X (gfx1250) — compile-verified, matrix operand order checked by reading, not run on hardware
//
#include <hip/hip_runtime.h>
#include <hip/hip_fp16.h>
#include <math.h>

typedef _Float16 v16h __attribute__((ext_vector_type(16)));
typedef _Float16 v8h  __attribute__((ext_vector_type(8)));
typedef float    v8f  __attribute__((ext_vector_type(8)));

#define S_DIM   128
#define V_DIM   32
#define L2_DIM  16
#define D_DIM   400
#define TEMB    128
#define NRBF    16
#define NTYPES  9
#define HID     64
#define GTOT    480
#define EDGEDP  416           // edge feature dim padded to 13 K-chunks of 32
#define NNODES  10000
#define NEDGES  200000
#define NT_TILES ((NEDGES + 15) / 16)

// ---------------- workspace layout (bytes) ----------------
#define OFF_SORT   0u            // NEDGES ints
#define OFF_CNT    800000u       // 16 ints
#define OFF_FILL   800064u       // 16 ints
#define OFF_W1B    800128u       // 9*13*4*512 halfs
#define OFF_W2B    1279360u      // 9*2*30*512
#define OFF_WS0B   1832320u      // 9*4*8*512
#define OFF_WSVB   2127232u      // 9*4*2*512
#define OFF_WSQB   2200960u      // 9*4*1*512
#define OFF_WVSB   2237824u      // 9*1*8*512
#define OFF_MIXB   2311552u      // 5*9*1*2*512
#define OFF_WQQB   2403712u      // 9*1*1*512
#define OFF_PSB    2412928u      // 1*4*8*512
#define OFF_WADAB  2445696u      // 1*4*21*512

// ---------------- WMMA helpers ----------------
__device__ inline v8f wmma16(v16h a, v16h b, v8f c) {
  return __builtin_amdgcn_wmma_f32_16x16x32_f16(false, a, false, b, (short)0, c,
                                                false, false);
}

// A-fragment (16x32 f16) from LDS; rows contiguous with stride rowStride halfs.
// Lane l: row = l%16; halves = K {sel*8..sel*8+7, 16+sel*8..16+sel*8+7}.
__device__ inline v16h load_a(const _Float16* base, int rowStride, int kOff, int lane) {
  const int row = lane & 15, sel = lane >> 4;
  const _Float16* p = base + row * rowStride + kOff + sel * 8;
  v8h lo = *(const v8h*)p;
  v8h hi = *(const v8h*)(p + 16);
  v16h r;
#pragma unroll
  for (int i = 0; i < 8; i++) { r[i] = lo[i]; r[i + 8] = hi[i]; }
  return r;
}

// B-fragment: pre-packed so each lane reads 16 contiguous halfs (32B).
__device__ inline v16h load_b(const _Float16* g, int fragIdx, int lane) {
  return *(const v16h*)(g + (size_t)fragIdx * 512 + lane * 16);
}

__device__ inline float sel3(float a, float b, float c, int i) {
  return i == 0 ? a : (i == 1 ? b : c);
}

// ---------------- utility kernels ----------------
__global__ void zero_f32_kernel(float* p, int n) {
  for (int i = blockIdx.x * blockDim.x + threadIdx.x; i < n; i += gridDim.x * blockDim.x)
    p[i] = 0.f;
}
__global__ void zero_i32_kernel(int* p, int n) {
  for (int i = blockIdx.x * blockDim.x + threadIdx.x; i < n; i += gridDim.x * blockDim.x)
    p[i] = 0;
}
__global__ void hist_kernel(const int* et, int* cnt) {
  __shared__ int loc[16];
  if (threadIdx.x < 16) loc[threadIdx.x] = 0;
  __syncthreads();
  for (int e = blockIdx.x * blockDim.x + threadIdx.x; e < NEDGES; e += gridDim.x * blockDim.x)
    atomicAdd(&loc[et[e]], 1);
  __syncthreads();
  if (threadIdx.x < NTYPES) atomicAdd(&cnt[threadIdx.x], loc[threadIdx.x]);
}
__global__ void scan_kernel(const int* cnt, int* fill) {
  if (threadIdx.x == 0) {
    int acc = 0;
    for (int t = 0; t < NTYPES; t++) { fill[t] = acc; acc += cnt[t]; }
  }
}
__global__ void scatter_kernel(const int* et, int* fill, int* sorted) {
  for (int e = blockIdx.x * blockDim.x + threadIdx.x; e < NEDGES; e += gridDim.x * blockDim.x) {
    int pos = atomicAdd(&fill[et[e]], 1);
    sorted[pos] = e;
  }
}

// Re-pack f32 weights [T][K][N] into f16 B-fragment layout [t][kc][nt][lane][16].
__global__ void pack_b_kernel(const float* src, _Float16* dst,
                              int T, int K, int N, int KC, int NT) {
  int total = T * KC * NT * 512;
  for (int gid = blockIdx.x * blockDim.x + threadIdx.x; gid < total;
       gid += gridDim.x * blockDim.x) {
    int hh   = gid & 15;
    int lane = (gid >> 4) & 31;
    int rest = gid >> 9;
    int nt = rest % NT; rest /= NT;
    int kc = rest % KC;
    int t  = rest / KC;
    int halfsel = lane >> 4;
    int k = kc * 32 + ((hh < 8) ? (halfsel * 8 + hh) : (16 + halfsel * 8 + (hh - 8)));
    int n = nt * 16 + (lane & 15);
    float v = (k < K && n < N) ? src[((size_t)t * K + k) * N + n] : 0.f;
    dst[gid] = (_Float16)v;
  }
}

// ---------------- edge (message) kernel ----------------
struct EdgeSmem {
  _Float16 ex[16 * EDGEDP];   // edge_x f16: [rbf16 | s_src128 | s_dst128 | temb128 | pad16]
  _Float16 Hh[16 * HID];      // MLP hidden
  _Float16 dotA[16 * 32];     // dot(vo_src, u)  (A for Wvs GEMM)
  _Float16 Aq[80 * 32];       // q_src rows: comp*16+edge, cols v (16..31 zero)
  _Float16 stA[48 * 32];      // reusable mix A: rows comp*16+edge, cols v
  float vo[16][32][3];
  float ve[16][32][3];
  float u[16][3];
  float Y2[16][5];
  int   srcn[16];
  int   dnode[16];
  int   etyp[16];
  int   valid[16];
  int   tmask;
};

__global__ __launch_bounds__(32) void edge_kernel(
    const float* __restrict__ h, const float* __restrict__ coords,
    const float* __restrict__ t_emb,
    const float* __restrict__ b1, const float* __restrict__ b2,
    const int* __restrict__ edge_index, const int* __restrict__ edge_type,
    const int* __restrict__ sorted_eid,
    const _Float16* __restrict__ W1B, const _Float16* __restrict__ W2B,
    const _Float16* __restrict__ Ws0B, const _Float16* __restrict__ WsvB,
    const _Float16* __restrict__ WsqB, const _Float16* __restrict__ WvsB,
    const _Float16* __restrict__ WmixB, const _Float16* __restrict__ WqqB,
    float* __restrict__ agg) {
  __shared__ EdgeSmem sm;
  const int lane = threadIdx.x;
  const int tile = blockIdx.x;
  const int ln = lane & 15, sel = lane >> 4;

  // ---- phase 0: per-edge meta + geometry (lanes 0..15) ----
  if (lane < 16) {
    int eidx = tile * 16 + lane;
    int ok = eidx < NEDGES;
    int e = sorted_eid[ok ? eidx : (NEDGES - 1)];
    int s = edge_index[e];
    int d = edge_index[NEDGES + e];
    sm.srcn[lane] = s; sm.dnode[lane] = d;
    sm.etyp[lane] = edge_type[e]; sm.valid[lane] = ok;
    float dx = coords[d * 3 + 0] - coords[s * 3 + 0];
    float dy = coords[d * 3 + 1] - coords[s * 3 + 1];
    float dz = coords[d * 3 + 2] - coords[s * 3 + 2];
    float dist = sqrtf(dx * dx + dy * dy + dz * dz);
    float inv = 1.0f / (dist + 1e-8f);
    float ux = dx * inv, uy = dy * inv, uz = dz * inv;
    sm.u[lane][0] = ux; sm.u[lane][1] = uy; sm.u[lane][2] = uz;
    const float s3 = 1.7320508075688772f;
    sm.Y2[lane][0] = s3 * ux * uy;
    sm.Y2[lane][1] = s3 * uy * uz;
    sm.Y2[lane][2] = 0.5f * (3.f * uz * uz - 1.f);
    sm.Y2[lane][3] = s3 * ux * uz;
    sm.Y2[lane][4] = 0.5f * s3 * (ux * ux - uy * uy);
    for (int c = 0; c < NRBF; c++) {
      float ctr = (10.0f / 15.0f) * (float)c;
      float z = (dist - ctr) * 1.6f;  // width = 0.625
      sm.ex[lane * EDGEDP + c] = (_Float16)__expf(-z * z);
    }
    for (int c = 400; c < EDGEDP; c++) sm.ex[lane * EDGEDP + c] = (_Float16)0.f;
  }
  if (lane == 0) {
    int m = 0;
    for (int j = 0; j < 16; j++) if (sm.valid[j]) m |= 1 << sm.etyp[j];
    sm.tmask = m;
  }

  // ---- phase 1: cooperative staging of edge_x / vo / ve / q / dot ----
  for (int j = 0; j < 16; j++) {
    const float* ps_ = h + (size_t)sm.srcn[j] * D_DIM;
    const float* pd_ = h + (size_t)sm.dnode[j] * D_DIM;
    const float* pt_ = t_emb + (size_t)sm.dnode[j] * TEMB;
#pragma unroll
    for (int r = 0; r < 4; r++) {
      int c = r * 32 + lane;
      sm.ex[j * EDGEDP + 16 + c]  = (_Float16)ps_[c];
      sm.ex[j * EDGEDP + 144 + c] = (_Float16)pd_[c];
      sm.ex[j * EDGEDP + 272 + c] = (_Float16)pt_[c];
    }
#pragma unroll
    for (int i = 0; i < 3; i++) {
      sm.vo[j][lane][i] = ps_[S_DIM + 3 * lane + i];
      sm.ve[j][lane][i] = ps_[S_DIM + 3 * V_DIM + 3 * lane + i];
    }
  }
  for (int m = 0; m < 5; m++)
    for (int j = 0; j < 16; j++) {
      float qv = (lane < 16) ? h[(size_t)sm.srcn[j] * D_DIM + 320 + 5 * lane + m] : 0.f;
      sm.Aq[(m * 16 + j) * 32 + lane] = (_Float16)qv;
    }
  for (int j = 0; j < 16; j++) {
    float dd = sm.vo[j][lane][0] * sm.u[j][0] + sm.vo[j][lane][1] * sm.u[j][1] +
               sm.vo[j][lane][2] * sm.u[j][2];
    sm.dotA[j * 32 + lane] = (_Float16)dd;
  }

  // ---- phase 2: per-type WMMA pipeline ----
  int tm = sm.tmask;
  while (tm) {
    const int t = __ffs(tm) - 1;
    tm &= tm - 1;
    __builtin_prefetch(W1B + (size_t)t * 13 * 4 * 512, 0, 0);

    unsigned rmask = 0;
    for (int j = 0; j < 16; j++)
      if (sm.valid[j] && sm.etyp[j] == t) rmask |= 1u << j;

    // MLP layer 1: H = silu(edge_x @ W1[t] + b1[t])   (16x416 @ 416x64)
    for (int nt = 0; nt < 4; nt++) {
      float bias = b1[t * HID + nt * 16 + ln];
      v8f acc;
#pragma unroll
      for (int i = 0; i < 8; i++) acc[i] = bias;
      for (int kc = 0; kc < 13; kc++)
        acc = wmma16(load_a(sm.ex, EDGEDP, kc * 32, lane),
                     load_b(W1B, (t * 13 + kc) * 4 + nt, lane), acc);
#pragma unroll
      for (int i = 0; i < 8; i++) {
        float x = acc[i];
        float sv = x / (1.f + __expf(-x));
        sm.Hh[(i + 8 * sel) * HID + nt * 16 + ln] = (_Float16)sv;
      }
    }
    v16h hk0 = load_a(sm.Hh, HID, 0, lane);
    v16h hk1 = load_a(sm.Hh, HID, 32, lane);

    // gate tile nt of G = H @ W2[t] + b2[t]   (2 WMMAs each)
    auto gate = [&](int nt) -> v8f {
      float bias = b2[t * GTOT + nt * 16 + ln];
      v8f c;
#pragma unroll
      for (int i = 0; i < 8; i++) c[i] = bias;
      c = wmma16(hk0, load_b(W2B, (t * 2 + 0) * 30 + nt, lane), c);
      c = wmma16(hk1, load_b(W2B, (t * 2 + 1) * 30 + nt, lane), c);
      return c;
    };

    // msg_s = g1 * (s_src @ Ws0) + g4 * (dot(vo,u) @ Wvs)
    for (int n = 0; n < 8; n++) {
      v8f g1 = gate(n);
      v8f g4 = gate(11 + n);
      v8f S0 = {};
      for (int kc = 0; kc < 4; kc++)
        S0 = wmma16(load_a(sm.ex, EDGEDP, 16 + kc * 32, lane),
                    load_b(Ws0B, (t * 4 + kc) * 8 + n, lane), S0);
      v8f D4 = {};
      D4 = wmma16(load_a(sm.dotA, 32, 0, lane), load_b(WvsB, t * 8 + n, lane), D4);
#pragma unroll
      for (int i = 0; i < 8; i++) {
        int row = i + 8 * sel;
        if ((rmask >> row) & 1)
          atomicAdd(&agg[(size_t)sm.dnode[row] * D_DIM + n * 16 + ln],
                    g1[i] * S0[i] + g4[i] * D4[i]);
      }
    }

    // msg_vo += (g2 * (s_src @ Wsv)) * u
    for (int n = 0; n < 2; n++) {
      v8f g2 = gate(8 + n);
      v8f Sv = {};
      for (int kc = 0; kc < 4; kc++)
        Sv = wmma16(load_a(sm.ex, EDGEDP, 16 + kc * 32, lane),
                    load_b(WsvB, (t * 4 + kc) * 2 + n, lane), Sv);
#pragma unroll
      for (int i = 0; i < 8; i++) {
        int row = i + 8 * sel;
        if ((rmask >> row) & 1) {
          float v = g2[i] * Sv[i];
          float* base = &agg[(size_t)sm.dnode[row] * D_DIM + S_DIM + 3 * (n * 16 + ln)];
          atomicAdd(base + 0, v * sm.u[row][0]);
          atomicAdd(base + 1, v * sm.u[row][1]);
          atomicAdd(base + 2, v * sm.u[row][2]);
        }
      }
    }

    // msg_q += (g3 * (s_src @ Wsq)) * Y2
    {
      v8f g3 = gate(10);
      v8f Sq = {};
      for (int kc = 0; kc < 4; kc++)
        Sq = wmma16(load_a(sm.ex, EDGEDP, 16 + kc * 32, lane),
                    load_b(WsqB, t * 4 + kc, lane), Sq);
#pragma unroll
      for (int i = 0; i < 8; i++) {
        int row = i + 8 * sel;
        if ((rmask >> row) & 1) {
          float v = g3[i] * Sq[i];
          float* base = &agg[(size_t)sm.dnode[row] * D_DIM + 320 + 5 * ln];
          for (int m = 0; m < 5; m++) atomicAdd(base + m, v * sm.Y2[row][m]);
        }
      }
    }

    // 5 gated V->V mixes: rows = comp*16 + edge, cols = v; gate frag layout matches
    const int gt0_tab[5]  = {21, 25, 28, 19, 23};        // W6, W8, W10, W5, W7
    const int dsto_tab[5] = {S_DIM, S_DIM, S_DIM, S_DIM + 96, S_DIM + 96};
    for (int op = 0; op < 5; op++) {
      for (int r = 0; r < 48; r++) {
        int comp = r >> 4, j = r & 15;
        float ux0 = sm.u[j][0], uy0 = sm.u[j][1], uz0 = sm.u[j][2];
        int a1 = (comp + 1) % 3, a2 = (comp + 2) % 3;
        float ua1 = sel3(ux0, uy0, uz0, a1), ua2 = sel3(ux0, uy0, uz0, a2);
        float val;
        if (op == 0)      val = sm.vo[j][lane][comp];                                   // vo_s
        else if (op == 1) val = sm.ve[j][lane][a1] * ua2 - sm.ve[j][lane][a2] * ua1;    // cross(ve,u)
        else if (op == 2) {                                                             // Mvo
          float dd = (float)sm.dotA[j * 32 + lane];
          val = sel3(ux0, uy0, uz0, comp) * dd - sm.vo[j][lane][comp] * (1.f / 3.f);
        } else if (op == 3) val = sm.vo[j][lane][a1] * ua2 - sm.vo[j][lane][a2] * ua1;  // cross(vo,u)
        else              val = sm.ve[j][lane][comp];                                   // ve_s
        sm.stA[r * 32 + lane] = (_Float16)val;
      }
      for (int n = 0; n < 2; n++) {
        v8f gf = gate(gt0_tab[op] + n);
        for (int comp = 0; comp < 3; comp++) {
          v8f mf = {};
          mf = wmma16(load_a(sm.stA + comp * 16 * 32, 32, 0, lane),
                      load_b(WmixB, (op * NTYPES + t) * 2 + n, lane), mf);
#pragma unroll
          for (int i = 0; i < 8; i++) {
            int row = i + 8 * sel;
            if ((rmask >> row) & 1)
              atomicAdd(&agg[(size_t)sm.dnode[row] * D_DIM + dsto_tab[op] +
                             3 * (n * 16 + ln) + comp],
                        gf[i] * mf[i]);
          }
        }
      }
    }

    // msg_q += g9 * (q_src @ Wqq)
    {
      v8f g9 = gate(27);
      for (int m = 0; m < 5; m++) {
        v8f qf = {};
        qf = wmma16(load_a(sm.Aq + m * 16 * 32, 32, 0, lane),
                    load_b(WqqB, t, lane), qf);
#pragma unroll
        for (int i = 0; i < 8; i++) {
          int row = i + 8 * sel;
          if ((rmask >> row) & 1)
            atomicAdd(&agg[(size_t)sm.dnode[row] * D_DIM + 320 + 5 * ln + m],
                      g9[i] * qf[i]);
        }
      }
    }
  }
}

// ---------------- node update kernel ----------------
struct NodeSmem {
  _Float16 aggsh[16 * 128];
  _Float16 tembh[16 * 128];
  float ps[16 * 128];
  float ada[16 * 336];
  float vo2[16 * 96];
  float ve2[16 * 96];
  float q2[16 * 80];
};

__global__ __launch_bounds__(32) void node_kernel(
    const float* __restrict__ h, const float* __restrict__ t_emb,
    const float* __restrict__ Pvo, const float* __restrict__ Pve,
    const float* __restrict__ Pq, const float* __restrict__ bada,
    const _Float16* __restrict__ PsB, const _Float16* __restrict__ WadaB,
    float* __restrict__ out) {   // holds agg on entry, final features on exit
  __shared__ NodeSmem sm;
  const int lane = threadIdx.x, ln = lane & 15, sel = lane >> 4;
  const int n0 = blockIdx.x * 16;

  for (int j = 0; j < 16; j++) {
    int n = n0 + j;
    bool ok = n < NNODES;
#pragma unroll
    for (int r = 0; r < 4; r++) {
      int c = r * 32 + lane;
      sm.aggsh[j * 128 + c] = (_Float16)(ok ? out[(size_t)n * D_DIM + c] : 0.f);
      sm.tembh[j * 128 + c] = (_Float16)(ok ? t_emb[(size_t)n * TEMB + c] : 0.f);
    }
  }

  // ps = silu(agg_s @ Ps)
  for (int nt = 0; nt < 8; nt++) {
    v8f acc = {};
    for (int kc = 0; kc < 4; kc++)
      acc = wmma16(load_a(sm.aggsh, 128, kc * 32, lane),
                   load_b(PsB, kc * 8 + nt, lane), acc);
#pragma unroll
    for (int i = 0; i < 8; i++) {
      float x = acc[i];
      sm.ps[(i + 8 * sel) * 128 + nt * 16 + ln] = x / (1.f + __expf(-x));
    }
  }
  // ada = t_emb @ Wada + bada   (N = 336 = 21 tiles)
  for (int nt = 0; nt < 21; nt++) {
    float bias = bada[nt * 16 + ln];
    v8f acc;
#pragma unroll
    for (int i = 0; i < 8; i++) acc[i] = bias;
    for (int kc = 0; kc < 4; kc++)
      acc = wmma16(load_a(sm.tembh, 128, kc * 32, lane),
                   load_b(WadaB, kc * 21 + nt, lane), acc);
#pragma unroll
    for (int i = 0; i < 8; i++)
      sm.ada[(i + 8 * sel) * 336 + nt * 16 + ln] = acc[i];
  }

  // per-node scalar tail (lanes 0..15)
  if (lane < 16) {
    int j = lane, n = n0 + j;
    if (n < NNODES) {
      const float* hn = h + (size_t)n * D_DIM;
      const float* an = out + (size_t)n * D_DIM;
      float* on = out + (size_t)n * D_DIM;
      for (int w = 0; w < 32; w++)
        for (int i = 0; i < 3; i++) {
          float pv = 0.f, pe = 0.f;
          for (int v = 0; v < 32; v++) {
            pv += an[128 + 3 * v + i] * Pvo[v * 32 + w];
            pe += an[224 + 3 * v + i] * Pve[v * 32 + w];
          }
          sm.vo2[j * 96 + 3 * w + i] = hn[128 + 3 * w + i] + pv;
          sm.ve2[j * 96 + 3 * w + i] = hn[224 + 3 * w + i] + pe;
        }
      for (int w = 0; w < 16; w++)
        for (int m = 0; m < 5; m++) {
          float pq = 0.f;
          for (int v = 0; v < 16; v++) pq += an[320 + 5 * v + m] * Pq[v * 16 + w];
          sm.q2[j * 80 + 5 * w + m] = hn[320 + 5 * w + m] + pq;
        }
      float mu = 0.f;
      for (int c = 0; c < 128; c++) mu += hn[c] + sm.ps[j * 128 + c];
      mu *= (1.f / 128.f);
      float var = 0.f;
      for (int c = 0; c < 128; c++) {
        float x = hn[c] + sm.ps[j * 128 + c] - mu;
        var += x * x;
      }
      float inv = rsqrtf(var * (1.f / 128.f) + 1e-6f);
      float sv = 0.f, se = 0.f, sq = 0.f;
      for (int c = 0; c < 96; c++) {
        sv += sm.vo2[j * 96 + c] * sm.vo2[j * 96 + c];
        se += sm.ve2[j * 96 + c] * sm.ve2[j * 96 + c];
      }
      for (int c = 0; c < 80; c++) sq += sm.q2[j * 80 + c] * sm.q2[j * 80 + c];
      float rv = rsqrtf(sv * (1.f / 96.f) + 1e-6f);
      float re = rsqrtf(se * (1.f / 96.f) + 1e-6f);
      float rq = rsqrtf(sq * (1.f / 80.f) + 1e-6f);
      const float* ad = &sm.ada[j * 336];
      for (int c = 0; c < 128; c++) {
        float x = hn[c] + sm.ps[j * 128 + c];
        on[c] = (x - mu) * inv * (1.f + ad[c]) + ad[128 + c];
      }
      for (int w = 0; w < 32; w++) {
        float gv = 1.f + ad[256 + w], ge = 1.f + ad[288 + w];
        for (int i = 0; i < 3; i++) {
          on[128 + 3 * w + i] = sm.vo2[j * 96 + 3 * w + i] * rv * gv;
          on[224 + 3 * w + i] = sm.ve2[j * 96 + 3 * w + i] * re * ge;
        }
      }
      for (int w = 0; w < 16; w++) {
        float gq = 1.f + ad[320 + w];
        for (int m = 0; m < 5; m++)
          on[320 + 5 * w + m] = sm.q2[j * 80 + 5 * w + m] * rq * gq;
      }
    }
  }
}

// ---------------- host-side launch ----------------
extern "C" void kernel_launch(void* const* d_in, const int* in_sizes, int n_in,
                              void* d_out, int out_size, void* d_ws, size_t ws_size,
                              hipStream_t stream) {
  (void)in_sizes; (void)n_in; (void)out_size; (void)ws_size;
  const float* h      = (const float*)d_in[0];
  const float* coords = (const float*)d_in[1];
  const float* t_emb  = (const float*)d_in[2];
  const float* W1     = (const float*)d_in[3];
  const float* b1     = (const float*)d_in[4];
  const float* W2     = (const float*)d_in[5];
  const float* b2     = (const float*)d_in[6];
  const float* Ws0    = (const float*)d_in[7];
  const float* Wsv    = (const float*)d_in[8];
  const float* Wsq    = (const float*)d_in[9];
  const float* Wvs    = (const float*)d_in[10];
  const float* W5     = (const float*)d_in[11];
  const float* W6     = (const float*)d_in[12];
  const float* W7     = (const float*)d_in[13];
  const float* W8     = (const float*)d_in[14];
  const float* Wqq    = (const float*)d_in[15];
  const float* W10    = (const float*)d_in[16];
  const float* Ps     = (const float*)d_in[17];
  const float* Pvo    = (const float*)d_in[18];
  const float* Pve    = (const float*)d_in[19];
  const float* Pq     = (const float*)d_in[20];
  const float* Wada   = (const float*)d_in[21];
  const float* bada   = (const float*)d_in[22];
  const int* edge_index = (const int*)d_in[23];
  const int* edge_type  = (const int*)d_in[24];
  float* out = (float*)d_out;

  char* ws = (char*)d_ws;
  int*       sorted = (int*)(ws + OFF_SORT);
  int*       cnt    = (int*)(ws + OFF_CNT);
  int*       fill   = (int*)(ws + OFF_FILL);
  _Float16*  W1B    = (_Float16*)(ws + OFF_W1B);
  _Float16*  W2B    = (_Float16*)(ws + OFF_W2B);
  _Float16*  Ws0B   = (_Float16*)(ws + OFF_WS0B);
  _Float16*  WsvB   = (_Float16*)(ws + OFF_WSVB);
  _Float16*  WsqB   = (_Float16*)(ws + OFF_WSQB);
  _Float16*  WvsB   = (_Float16*)(ws + OFF_WVSB);
  _Float16*  WmixB  = (_Float16*)(ws + OFF_MIXB);
  _Float16*  WqqB   = (_Float16*)(ws + OFF_WQQB);
  _Float16*  PsB    = (_Float16*)(ws + OFF_PSB);
  _Float16*  WadaB  = (_Float16*)(ws + OFF_WADAB);

  // 1) init accumulator (d_out doubles as agg buffer) + counters
  zero_f32_kernel<<<2048, 256, 0, stream>>>(out, NNODES * D_DIM);
  zero_i32_kernel<<<1, 64, 0, stream>>>(cnt, 32);

  // 2) pack all weights into WMMA B-fragment f16 layout
  auto packs = [&](const float* src, _Float16* dst, int T, int K, int N, int KC, int NT) {
    int total = T * KC * NT * 512;
    pack_b_kernel<<<(total + 255) / 256, 256, 0, stream>>>(src, dst, T, K, N, KC, NT);
  };
  packs(W1,  W1B,  NTYPES, 400, 64, 13, 4);
  packs(W2,  W2B,  NTYPES, 64, 480, 2, 30);
  packs(Ws0, Ws0B, NTYPES, 128, 128, 4, 8);
  packs(Wsv, WsvB, NTYPES, 128, 32, 4, 2);
  packs(Wsq, WsqB, NTYPES, 128, 16, 4, 1);
  packs(Wvs, WvsB, NTYPES, 32, 128, 1, 8);
  packs(W6,  WmixB + 0 * NTYPES * 2 * 512, NTYPES, 32, 32, 1, 2);
  packs(W8,  WmixB + 1 * NTYPES * 2 * 512, NTYPES, 32, 32, 1, 2);
  packs(W10, WmixB + 2 * NTYPES * 2 * 512, NTYPES, 32, 32, 1, 2);
  packs(W5,  WmixB + 3 * NTYPES * 2 * 512, NTYPES, 32, 32, 1, 2);
  packs(W7,  WmixB + 4 * NTYPES * 2 * 512, NTYPES, 32, 32, 1, 2);
  packs(Wqq, WqqB, NTYPES, 16, 16, 1, 1);
  packs(Ps,  PsB,  1, 128, 128, 4, 8);
  packs(Wada, WadaB, 1, 128, 336, 4, 21);

  // 3) bin edges by type so WMMA tiles are type-uniform
  hist_kernel<<<256, 256, 0, stream>>>(edge_type, cnt);
  scan_kernel<<<1, 32, 0, stream>>>(cnt, fill);
  scatter_kernel<<<256, 256, 0, stream>>>(edge_type, fill, sorted);

  // 4) edge messages (one wave32 per 16-edge tile)
  edge_kernel<<<NT_TILES, 32, 0, stream>>>(
      h, coords, t_emb, b1, b2, edge_index, edge_type, sorted,
      W1B, W2B, Ws0B, WsvB, WsqB, WvsB, WmixB, WqqB, out);

  // 5) node update (one wave32 per 16 nodes, in place on d_out)
  node_kernel<<<(NNODES + 15) / 16, 32, 0, stream>>>(
      h, t_emb, Pvo, Pve, Pq, bada, PsB, WadaB, out);
}
